// DepthNet_54082228191759
// MI455X (gfx1250) — matrix operand order repaired and checked by reading, not compiled, not run on hardware
//
#include <hip/hip_runtime.h>
#include <hip/hip_bf16.h>
#include <math.h>

// Problem constants (from reference setup_inputs)
static constexpr int B = 2, V = 3, C = 32, D = 48, H = 128, W = 160;
static constexpr int HW = H * W;
static constexpr int NWT = W / 16;          // 16-pixel tiles along W

typedef __attribute__((ext_vector_type(16))) __bf16 v16bf;
typedef __attribute__((ext_vector_type(8)))  float  v8f;
typedef __attribute__((ext_vector_type(4)))  float  fx4;

union AV { v16bf v; fx4 f[2]; };

// ---------------------------------------------------------------------------
// Kernel 0: per-batch projection matrices (src_proj @ inv(ref_proj)) and the
// bf16 tap-major weight table wtab[27][32].
// ---------------------------------------------------------------------------
__global__ void k_setup(const float* __restrict__ proj,
                        const float* __restrict__ regw,
                        float* __restrict__ rt,
                        __bf16* __restrict__ wtab) {
  int tid = threadIdx.x;

  // reorder reg_w (1,C,3,3,3) -> wtab[t*32+c], t = kd*9+kh*3+kw
  if (tid < 27 * 32) {
    int t = tid >> 5;
    int c = tid & 31;
    wtab[t * 32 + c] = (__bf16)regw[c * 27 + t];
  }

  if (tid < B) {
    int b = tid;
    const float* refp = proj + (size_t)(b * V + 0) * 16;
    // Gauss-Jordan inverse of ref_proj with partial pivoting
    float a[4][8];
    for (int i = 0; i < 4; ++i)
      for (int j = 0; j < 4; ++j) {
        a[i][j] = refp[i * 4 + j];
        a[i][4 + j] = (i == j) ? 1.0f : 0.0f;
      }
    for (int col = 0; col < 4; ++col) {
      int piv = col;
      float best = fabsf(a[col][col]);
      for (int r = col + 1; r < 4; ++r) {
        float vv = fabsf(a[r][col]);
        if (vv > best) { best = vv; piv = r; }
      }
      if (piv != col)
        for (int j = 0; j < 8; ++j) {
          float t2 = a[col][j]; a[col][j] = a[piv][j]; a[piv][j] = t2;
        }
      float ip = 1.0f / a[col][col];
      for (int j = 0; j < 8; ++j) a[col][j] *= ip;
      for (int r = 0; r < 4; ++r)
        if (r != col) {
          float f = a[r][col];
          for (int j = 0; j < 8; ++j) a[r][j] -= f * a[col][j];
        }
    }
    float inv[16];
    for (int i = 0; i < 4; ++i)
      for (int j = 0; j < 4; ++j) inv[i * 4 + j] = a[i][4 + j];

    for (int v = 1; v < V; ++v) {
      const float* sp = proj + (size_t)(b * V + v) * 16;
      float Mv[16];
      for (int i = 0; i < 4; ++i)
        for (int j = 0; j < 4; ++j) {
          float s = 0.0f;
          for (int k = 0; k < 4; ++k) s += sp[i * 4 + k] * inv[k * 4 + j];
          Mv[i * 4 + j] = s;
        }
      float* e = rt + (size_t)(b * (V - 1) + (v - 1)) * 12;
      e[0] = Mv[0];  e[1] = Mv[1];  e[2] = Mv[2];
      e[3] = Mv[4];  e[4] = Mv[5];  e[5] = Mv[6];
      e[6] = Mv[8];  e[7] = Mv[9];  e[8] = Mv[10];
      e[9] = Mv[3];  e[10] = Mv[7]; e[11] = Mv[11];
    }
  }
}

// ---------------------------------------------------------------------------
// Kernel 1: fused homography warp + variance; writes bf16 channel-last volume
// var[(((b*D+d)*H+h)*W+w)*32 + c].  One thread per (b,d,h,w).
// ---------------------------------------------------------------------------
__global__ void k_warp_var(const float* __restrict__ feats,
                           const float* __restrict__ dvals,
                           const float* __restrict__ rt,
                           __bf16* __restrict__ var) {
  int i = blockIdx.x * blockDim.x + threadIdx.x;  // exact grid, no tail
  int w = i % W;
  int h = (i / W) % H;
  int d = (i / HW) % D;
  int b = i / (HW * D);

  float wf = (float)w, hf = (float)h;
  float depth = dvals[b * D + d];

  float wgt[V - 1][4];
  int   off[V - 1][4];
  const float* vb[V - 1];

#pragma unroll
  for (int v = 1; v < V; ++v) {
    const int vi = v - 1;
    const float* e = rt + (size_t)(b * (V - 1) + vi) * 12;
    float rx = e[0] * wf + e[1] * hf + e[2];
    float ry = e[3] * wf + e[4] * hf + e[5];
    float rz = e[6] * wf + e[7] * hf + e[8];
    float pxn = rx * depth + e[9];
    float pyn = ry * depth + e[10];
    float pz  = rz * depth + e[11];
    pz = (fabsf(pz) < 1e-6f) ? 1e-6f : pz;   // matches jnp.where(|z|<eps, eps, z)
    float px = pxn / pz, py = pyn / pz;

    float x0f = floorf(px), y0f = floorf(py);
    float x1f = x0f + 1.0f, y1f = y0f + 1.0f;
    float wx1 = px - x0f, wx0 = 1.0f - wx1;
    float wy1 = py - y0f, wy0 = 1.0f - wy1;

    float vx0 = (x0f >= 0.0f && x0f <= (float)(W - 1)) ? 1.0f : 0.0f;
    float vx1 = (x1f >= 0.0f && x1f <= (float)(W - 1)) ? 1.0f : 0.0f;
    float vy0 = (y0f >= 0.0f && y0f <= (float)(H - 1)) ? 1.0f : 0.0f;
    float vy1 = (y1f >= 0.0f && y1f <= (float)(H - 1)) ? 1.0f : 0.0f;

    int x0 = (int)x0f, y0 = (int)y0f;
    int xc0 = min(max(x0, 0), W - 1);
    int xc1 = min(max(x0 + 1, 0), W - 1);
    int yc0 = min(max(y0, 0), H - 1);
    int yc1 = min(max(y0 + 1, 0), H - 1);

    off[vi][0] = yc0 * W + xc0;  wgt[vi][0] = wx0 * wy0 * vx0 * vy0;
    off[vi][1] = yc0 * W + xc1;  wgt[vi][1] = wx1 * wy0 * vx1 * vy0;
    off[vi][2] = yc1 * W + xc0;  wgt[vi][2] = wx0 * wy1 * vx0 * vy1;
    off[vi][3] = yc1 * W + xc1;  wgt[vi][3] = wx1 * wy1 * vx1 * vy1;
    vb[vi] = feats + ((size_t)v * B + b) * C * HW;
  }

  const float* rf = feats + (size_t)b * C * HW + h * W + w;  // ref view (v=0)

  union { __bf16 hb[32]; fx4 f[4]; } ob;
  constexpr float invV = 1.0f / (float)V;

#pragma unroll
  for (int c = 0; c < C; ++c) {
    float s = rf[(size_t)c * HW];
    float q = s * s;
#pragma unroll
    for (int vi = 0; vi < V - 1; ++vi) {
      const float* fb = vb[vi] + (size_t)c * HW;
      float val = fb[off[vi][0]] * wgt[vi][0] + fb[off[vi][1]] * wgt[vi][1] +
                  fb[off[vi][2]] * wgt[vi][2] + fb[off[vi][3]] * wgt[vi][3];
      s += val;
      q += val * val;
    }
    float m = s * invV;
    ob.hb[c] = (__bf16)(q * invV - m * m);
  }

  fx4* op = (fx4*)(var + (size_t)i * 32);
  op[0] = ob.f[0]; op[1] = ob.f[1]; op[2] = ob.f[2]; op[3] = ob.f[3];
}

// ---------------------------------------------------------------------------
// Kernel 2: 3D conv (C=32 -> 1, 3x3x3, pad 1) with WMMA bf16.
// One wave per 16-pixel tile along W.  A = 16 pixels x 32 channels (bf16),
// B = tap weights replicated across 16 columns, 27 taps accumulated into one
// f32 fragment.  The 1.7 KB tap table is staged in LDS once per workgroup and
// read with ds_load_b128 (lanes 0-15 broadcast one 16B chunk, lanes 16-31 the
// other), eliminating ~27 KB of redundant global weight traffic per tile.
// EXEC is all-ones at every WMMA (exact grid, uniform tap skips, per-lane
// register zeroing at W boundaries with EXEC restored before the WMMA).
// ---------------------------------------------------------------------------
__global__ void k_cost_wmma(const __bf16* __restrict__ var,
                            const __bf16* __restrict__ wtab,
                            const float* __restrict__ regb,
                            float* __restrict__ cost) {
  __shared__ __align__(16) __bf16 swt[27 * 32];   // 1728 B tap table

  int tidb = threadIdx.x;
  // cooperative stage: 432 dwords, 256 threads
  {
    const unsigned int* src = (const unsigned int*)wtab;
    unsigned int* dst = (unsigned int*)swt;
    for (int j = tidb; j < 27 * 32 / 2; j += 256) dst[j] = src[j];
  }
  __syncthreads();

  int tid  = blockIdx.x * blockDim.x + tidb;
  int lane = tidb & 31;
  int wave = tid >> 5;

  int wt = wave % NWT;
  int h  = (wave / NWT) % H;
  int d  = (wave / (NWT * H)) % D;
  int b  =  wave / (NWT * H * D);
  int w0 = wt * 16;
  int m   = lane & 15;   // A row (output pixel within tile)
  int sel = lane >> 4;   // K-chunk select for the half-lane

  v8f acc = {};

#pragma unroll
  for (int kd = 0; kd < 3; ++kd) {
    int dd = d + kd - 1;
    if (dd < 0 || dd >= D) continue;            // uniform skip
#pragma unroll
    for (int kh = 0; kh < 3; ++kh) {
      int hh = h + kh - 1;
      if (hh < 0 || hh >= H) continue;          // uniform skip
      const __bf16* rowb = var + ((((size_t)b * D + dd) * H + hh) * (size_t)W) * 32;
      __builtin_prefetch(rowb + (size_t)w0 * 32, 0, 1);   // global_prefetch_b8
#pragma unroll
      for (int kw = 0; kw < 3; ++kw) {
        int ww = w0 + m + kw - 1;
        AV a;
        if (ww >= 0 && ww < W) {
          const fx4* p = (const fx4*)(rowb + (size_t)ww * 32);
          a.f[0] = p[sel];        // K 0-7  (lanes 0-15) / 8-15  (lanes 16-31)
          a.f[1] = p[2 + sel];    // K 16-23            / 24-31
        } else {
          fx4 z = {};
          a.f[0] = z; a.f[1] = z; // zero A rows outside W, keep EXEC full
        }
        int tap = (kd * 3 + kh) * 3 + kw;
        const fx4* wp = (const fx4*)(swt + tap * 32);     // LDS broadcast
        AV bf;
        bf.f[0] = wp[sel];          // lanes 0-15: K 0-7 ; lanes 16-31: K 8-15
        bf.f[1] = wp[2 + sel];      // K 16-23 / 24-31
        acc = __builtin_amdgcn_wmma_f32_16x16x32_bf16(
            /*neg_a=*/false, a.v, /*neg_b=*/false, bf.v,
            /*c_mod=*/(short)0, acc, /*reuse_a=*/false, /*reuse_b=*/false);
      }
    }
  }

  float bias = regb[0];
  union { v8f v; fx4 f[2]; } cu;
  cu.v = acc;
  // D layout: lane 0 holds M=0..7 (col 0), lane 16 holds M=8..15 (col 0)
  if (lane == 0 || lane == 16) {
    fx4 r0 = cu.f[0] + bias;
    fx4 r1 = cu.f[1] + bias;
    float* dst = cost + (((size_t)b * D + d) * H + h) * (size_t)W + w0 + (lane >> 1);
    ((fx4*)dst)[0] = r0;
    ((fx4*)dst)[1] = r1;
  }
}

// ---------------------------------------------------------------------------
// Kernel 3: softmax over D, expected depth, window-4 confidence.
// One thread per (b,h,w); all 48 exponentials kept in registers.
// ---------------------------------------------------------------------------
__global__ void k_softmax_depth(const float* __restrict__ cost,
                                const float* __restrict__ dvals,
                                float* __restrict__ out_depth,
                                float* __restrict__ out_conf) {
  int n = blockIdx.x * blockDim.x + threadIdx.x;   // exact grid
  int w = n % W;
  int h = (n / W) % H;
  int b = n / HW;

  const float* cp = cost + (size_t)b * D * HW + h * W + w;

  float e[D];
  float mx = -3.402823466e38f;
#pragma unroll
  for (int d = 0; d < D; ++d) {
    e[d] = cp[(size_t)d * HW];
    mx = fmaxf(mx, e[d]);
  }
  float s = 0.0f;
#pragma unroll
  for (int d = 0; d < D; ++d) {
    e[d] = __expf(e[d] - mx);
    s += e[d];
  }
  float inv = 1.0f / s;
  float depth = 0.0f, di = 0.0f;
#pragma unroll
  for (int d = 0; d < D; ++d) {
    float p = e[d] * inv;
    depth += p * dvals[b * D + d];
    di    += p * (float)d;
  }
  int idx = (int)di;
  idx = min(max(idx, 0), D - 1);
  float conf = 0.0f;
#pragma unroll
  for (int d = 0; d < D; ++d) {
    bool inwin = (d >= idx - 1) && (d <= idx + 2);
    conf += inwin ? e[d] : 0.0f;
  }
  conf *= inv;

  out_depth[n] = depth;
  out_conf[n]  = conf;
}

// ---------------------------------------------------------------------------
extern "C" void kernel_launch(void* const* d_in, const int* in_sizes, int n_in,
                              void* d_out, int out_size, void* d_ws, size_t ws_size,
                              hipStream_t stream) {
  (void)in_sizes; (void)n_in; (void)out_size; (void)ws_size;

  const float* feats = (const float*)d_in[0];  // (V,B,C,H,W)
  const float* proj  = (const float*)d_in[1];  // (B,V,4,4)
  const float* dvals = (const float*)d_in[2];  // (B,D)
  /* d_in[3] = num_depth (int scalar), value known at compile time */
  const float* regw  = (const float*)d_in[4];  // (1,C,3,3,3)
  const float* regb  = (const float*)d_in[5];  // (1,)

  float* out = (float*)d_out;                  // depth (B,H,W) then conf (B,H,W)

  // Workspace layout
  char* ws = (char*)d_ws;
  float*   rt   = (float*)(ws + 0);            // 2*2*12 floats (rot|trans)
  __bf16*  wtab = (__bf16*)(ws + 256);         // 27*32 bf16 weight table
  __bf16*  var  = (__bf16*)(ws + 4096);        // B*D*H*W*32 bf16 (~126 MB)
  size_t var_bytes = (size_t)B * D * H * W * C * sizeof(__bf16);
  float*   cost = (float*)(ws + 4096 + var_bytes);  // B*D*H*W floats

  k_setup<<<1, 896, 0, stream>>>(proj, regw, rt, wtab);

  int n1 = B * D * H * W;                      // 1,966,080 (divisible by 256)
  k_warp_var<<<n1 / 256, 256, 0, stream>>>(feats, dvals, rt, var);

  int nwaves = B * D * H * NWT;                // 122,880 waves
  k_cost_wmma<<<(nwaves * 32) / 256, 256, 0, stream>>>(var, wtab, regb, cost);

  int n3 = B * H * W;                          // 40,960 (divisible by 256)
  k_softmax_depth<<<n3 / 256, 256, 0, stream>>>(cost, dvals, out, out + n3);
}
